// VqVAE_33414845563510
// MI455X (gfx1250) — compile-verified, matrix operand order checked by reading, not run on hardware
//
#include <hip/hip_runtime.h>

// ---------------------------------------------------------------------------
// VQ-VAE forward for MI455X (gfx1250): 100 independent per-variable MLPs.
// bf16 WMMA GEMMs (v_wmma_f32_16x16x32_bf16) with f32 accumulate.
// Weights stream f32 from HBM (NT hint), converted to bf16 into LDS.
// Activations ping-pong as bf16; A-tiles staged via async global->LDS copies.
// ---------------------------------------------------------------------------

#define NVAR   100
#define INDIM  99
#define BATCH  256
#define DIMQ   64
#define KCODE  128

typedef __attribute__((ext_vector_type(16))) __bf16          v16bf;
typedef __attribute__((ext_vector_type(16))) unsigned short  v16us;
typedef __attribute__((ext_vector_type(8)))  float           v8f;
typedef __attribute__((ext_vector_type(4)))  float           f4;

__device__ __forceinline__ unsigned short f32_to_bf16(float f) {
  unsigned int u = __builtin_bit_cast(unsigned int, f);
  unsigned int r = u + 0x7FFFu + ((u >> 16) & 1u);   // round-to-nearest-even
  return (unsigned short)(r >> 16);
}
__device__ __forceinline__ float bf16_to_f32(unsigned short h) {
  unsigned int u = ((unsigned int)h) << 16;
  return __builtin_bit_cast(float, u);
}

enum ActKind { ACT_SELU, ACT_SIGMOID };

template <ActKind A> __device__ __forceinline__ float apply_act(float x);
template <> __device__ __forceinline__ float apply_act<ACT_SELU>(float x) {
  const float kScale = 1.0507009873554805f;
  const float kAlpha = 1.6732632423543772f;
  return x > 0.0f ? kScale * x : kScale * kAlpha * (__expf(x) - 1.0f);
}
template <> __device__ __forceinline__ float apply_act<ACT_SIGMOID>(float x) {
  return 1.0f / (1.0f + __expf(-x));
}

// ---- A-tile staging: 256 rows x 32 K, bf16 into sA[row*40 + k] -------------
// bf16 activations: fully aligned 16B chunks -> async global->LDS copies.
__device__ __forceinline__ void stage_a(unsigned short* sA,
                                        const unsigned short* An, long rs,
                                        int k0, int din, bool fullK, int tid) {
  if (fullK) {                               // always true for bf16 layers
    #pragma unroll
    for (int r = 0; r < 4; ++r) {
      int e   = tid + r * 256;               // 0..1023
      int row = e >> 2;                      // 0..255
      int ch  = (e & 3) * 8;                 // K sub-chunk (8 halves = 16B)
      const unsigned short* src = An + (long)row * rs + k0 + ch;
      unsigned dst = (unsigned)(unsigned long long)&sA[row * 40 + ch];
      asm volatile("global_load_async_to_lds_b128 %0, %1, off"
                   :: "v"(dst), "v"(src) : "memory");
    }
  } else {                                   // not reached for bf16, keep safe
    #pragma unroll 4
    for (int r = 0; r < 32; ++r) {
      int e = tid + r * 256;
      int row = e >> 5, col = e & 31;
      int k = k0 + col;
      unsigned short v = 0;
      if (k < din) v = An[(long)row * rs + k];
      sA[row * 40 + col] = v;
    }
  }
}
// f32 activations (first layer, x is [B, NVAR, INDIM], rows not 16B aligned)
__device__ __forceinline__ void stage_a(unsigned short* sA,
                                        const float* An, long rs,
                                        int k0, int din, bool fullK, int tid) {
  if (fullK) {
    #pragma unroll 4
    for (int r = 0; r < 32; ++r) {
      int e = tid + r * 256;
      int row = e >> 5, col = e & 31;
      sA[row * 40 + col] = f32_to_bf16(An[(long)row * rs + k0 + col]);
    }
  } else {
    #pragma unroll 4
    for (int r = 0; r < 32; ++r) {
      int e = tid + r * 256;
      int row = e >> 5, col = e & 31;
      int k = k0 + col;
      unsigned short v = 0;
      if (k < din) v = f32_to_bf16(An[(long)row * rs + k]);
      sA[row * 40 + col] = v;
    }
  }
}

// ---------------------------------------------------------------------------
// Batched GEMM layer: Out[n] = act( A[n] (256 x din) * W[n] (din x dout) + b[n] )
// Block: 256 thr = 8 waves (4 M x 2 N); block tile 256x64; wave tile 64x32.
// Per K-step: 8x v_wmma_f32_16x16x32_bf16 per wave.
// ---------------------------------------------------------------------------
template <ActKind ACT, bool TRANS_OUT, typename TA>
__global__ __launch_bounds__(256) void gemm_layer(
    const TA* __restrict__ A, long a_n_stride, long a_row_stride,
    const float* __restrict__ W, const float* __restrict__ bias,
    void* __restrict__ Out, int din, int dout)
{
  __shared__ __align__(16) unsigned short sA[256 * 40];   // 20.0 KB
  __shared__ __align__(16) unsigned short sB[64 * 40];    //  5.0 KB

  const int n    = blockIdx.z;
  const int colB = blockIdx.x * 64;
  const int tid  = threadIdx.x;
  const int lane = tid & 31;
  const int w    = tid >> 5;
  const int wm   = w & 3;                 // rows wm*64
  const int wn   = w >> 2;                // cols wn*32

  const TA*    An = A + (long)n * a_n_stride;
  const float* Wn = W + (long)n * (long)din * (long)dout;
  const float* bn = bias + (long)n * (long)dout;

  v8f acc[4][2] = {};

  const int nkt = (din + 31) >> 5;
  const int fm  = lane & 15;              // fragment row (A) / col (B)
  const int kb  = (lane >> 4) * 8;        // fragment K sub-base

  for (int kt = 0; kt < nkt; ++kt) {
    const int  k0    = kt << 5;
    const bool fullK = (k0 + 32) <= din;

    // ---- A tile ----
    stage_a(sA, An, a_row_stride, k0, din, fullK, tid);

    // ---- W tile (transposed into sB[n_local*40 + k_local]) ----
    if (fullK && (colB + 64) <= dout) {
      #pragma unroll
      for (int r = 0; r < 2; ++r) {
        int e  = tid + r * 256;           // 0..511
        int kl = e >> 4;                  // 0..31
        int nc = (e & 15) * 4;            // 0..60
        const f4* wp = (const f4*)&Wn[(long)(k0 + kl) * dout + colB + nc];
        f4 wv = __builtin_nontemporal_load(wp);   // NT: streamed once, skip L2 residency
        sB[(nc + 0) * 40 + kl] = f32_to_bf16(wv.x);
        sB[(nc + 1) * 40 + kl] = f32_to_bf16(wv.y);
        sB[(nc + 2) * 40 + kl] = f32_to_bf16(wv.z);
        sB[(nc + 3) * 40 + kl] = f32_to_bf16(wv.w);
        if (kt + 1 < nkt)                 // prefetch next K-slab (global_prefetch_b8)
          __builtin_prefetch(&Wn[(long)(k0 + 32 + kl) * dout + colB + nc], 0, 1);
      }
    } else {
      #pragma unroll 2
      for (int r = 0; r < 8; ++r) {
        int e  = tid + r * 256;
        int kl = e >> 6, nl = e & 63;
        int k  = k0 + kl, c = colB + nl;
        unsigned short v = 0;
        if (k < din && c < dout) v = f32_to_bf16(Wn[(long)k * dout + c]);
        sB[nl * 40 + kl] = v;
      }
    }

    asm volatile("s_wait_asynccnt 0" ::: "memory");
    __syncthreads();

    // ---- fragments + WMMA ----
    v16us bfr[2];
    #pragma unroll
    for (int j = 0; j < 2; ++j) {
      const unsigned short* pb = &sB[(wn * 32 + j * 16 + fm) * 40 + kb];
      #pragma unroll
      for (int e2 = 0; e2 < 8; ++e2) { bfr[j][e2] = pb[e2]; bfr[j][8 + e2] = pb[16 + e2]; }
    }
    #pragma unroll
    for (int i = 0; i < 4; ++i) {
      v16us afr;
      const unsigned short* pa = &sA[(wm * 64 + i * 16 + fm) * 40 + kb];
      #pragma unroll
      for (int e2 = 0; e2 < 8; ++e2) { afr[e2] = pa[e2]; afr[8 + e2] = pa[16 + e2]; }
      #pragma unroll
      for (int j = 0; j < 2; ++j)
        acc[i][j] = __builtin_amdgcn_wmma_f32_16x16x32_bf16(
            false, __builtin_bit_cast(v16bf, afr),
            false, __builtin_bit_cast(v16bf, bfr[j]),
            (short)0, acc[i][j], false, false);
    }
    __syncthreads();
  }

  // ---- epilogue: bias + activation + store ----
  const int hi = lane >> 4;
  const int cn = lane & 15;
  #pragma unroll
  for (int i = 0; i < 4; ++i) {
    const int rowb = wm * 64 + i * 16 + hi * 8;
    #pragma unroll
    for (int j = 0; j < 2; ++j) {
      const int colb = colB + wn * 32 + j * 16 + cn;
      if (colb >= dout) continue;
      const float bv = bn[colb];
      #pragma unroll
      for (int v = 0; v < 8; ++v) {
        float r = apply_act<ACT>(acc[i][j][v] + bv);
        if (TRANS_OUT) {
          // final layer: f32, [batch, nvar, INDIM]
          ((float*)Out)[(long)(rowb + v) * (NVAR * INDIM) + (long)n * INDIM + colb] = r;
        } else {
          ((unsigned short*)Out)[((long)n * BATCH + rowb + v) * dout + colb] =
              f32_to_bf16(r);
        }
      }
    }
  }
}

// ---------------------------------------------------------------------------
// Vector quantization: per var, argmin_k ||h - c_k||^2, forward output = c_k.
// One block per variable (256 thr = 1 per batch sample), codebook in LDS.
// ---------------------------------------------------------------------------
__global__ __launch_bounds__(256) void vq_kernel(
    const unsigned short* __restrict__ Hin,   // [NVAR, BATCH, DIMQ] bf16
    const float* __restrict__ codebook,       // [NVAR, KCODE, DIMQ] f32
    unsigned short* __restrict__ Hout)        // [NVAR, BATCH, DIMQ] bf16
{
  __shared__ float cb[KCODE * DIMQ];          // 32 KB
  const int n   = blockIdx.x;
  const int tid = threadIdx.x;

  const float* cbn = codebook + (long)n * KCODE * DIMQ;
  for (int e = tid; e < KCODE * DIMQ; e += 256) cb[e] = cbn[e];
  __syncthreads();

  float h[DIMQ];
  const unsigned short* hp = Hin + ((long)n * BATCH + tid) * DIMQ;
  #pragma unroll
  for (int d = 0; d < DIMQ; ++d) h[d] = bf16_to_f32(hp[d]);

  int   best  = 0;
  float bestd = 3.4e38f;
  for (int c = 0; c < KCODE; ++c) {
    float s = 0.0f;
    #pragma unroll
    for (int d = 0; d < DIMQ; ++d) {
      float df = h[d] - cb[c * DIMQ + d];     // broadcast LDS read
      s = fmaf(df, df, s);
    }
    if (s < bestd) { bestd = s; best = c; }
  }

  unsigned short* op = Hout + ((long)n * BATCH + tid) * DIMQ;
  #pragma unroll
  for (int d = 0; d < DIMQ; ++d) op[d] = f32_to_bf16(cb[best * DIMQ + d]);
}

// ---------------------------------------------------------------------------
extern "C" void kernel_launch(void* const* d_in, const int* in_sizes, int n_in,
                              void* d_out, int out_size, void* d_ws, size_t ws_size,
                              hipStream_t stream) {
  (void)in_sizes; (void)n_in; (void)out_size; (void)ws_size;

  const float* x = (const float*)d_in[0];
  const float* Wp[10];
  const float* bp[10];
  for (int i = 0; i < 10; ++i) {
    Wp[i] = (const float*)d_in[1 + 2 * i];
    bp[i] = (const float*)d_in[2 + 2 * i];
  }
  const float* codebook = (const float*)d_in[21];

  // bf16 ping-pong activation buffers (max dim 1024)
  unsigned short* buf0 = (unsigned short*)d_ws;
  unsigned short* buf1 = buf0 + (size_t)NVAR * BATCH * 1024;

  dim3 blk(256);
  auto grid = [](int dout) { return dim3((dout + 63) / 64, 1, NVAR); };

  const int enc_din[5]  = { INDIM, 1024, 512, 256, 128 };
  const int enc_dout[5] = { 1024,  512,  256, 128, DIMQ };
  const int dec_din[5]  = { DIMQ, 128, 256, 512, 1024 };
  const int dec_dout[5] = { 128,  256, 512, 1024, INDIM };

  // ---------------- encoder ----------------
  gemm_layer<ACT_SELU, false><<<grid(enc_dout[0]), blk, 0, stream>>>(
      x, (long)INDIM, (long)(NVAR * INDIM),
      Wp[0], bp[0], buf0, enc_din[0], enc_dout[0]);

  unsigned short* src = buf0;
  unsigned short* dst = buf1;
  for (int l = 1; l < 5; ++l) {
    gemm_layer<ACT_SELU, false><<<grid(enc_dout[l]), blk, 0, stream>>>(
        src, (long)BATCH * enc_din[l], (long)enc_din[l],
        Wp[l], bp[l], dst, enc_din[l], enc_dout[l]);
    unsigned short* t = src; src = dst; dst = t;
  }

  // ---------------- vector quantization (forward == selected code) --------
  vq_kernel<<<dim3(NVAR), blk, 0, stream>>>(src, codebook, dst);
  { unsigned short* t = src; src = dst; dst = t; }

  // ---------------- decoder ----------------
  for (int l = 0; l < 4; ++l) {
    gemm_layer<ACT_SELU, false><<<grid(dec_dout[l]), blk, 0, stream>>>(
        src, (long)BATCH * dec_din[l], (long)dec_din[l],
        Wp[5 + l], bp[5 + l], dst, dec_din[l], dec_dout[l]);
    unsigned short* t = src; src = dst; dst = t;
  }
  gemm_layer<ACT_SIGMOID, true><<<grid(dec_dout[4]), blk, 0, stream>>>(
      src, (long)BATCH * dec_din[4], (long)dec_din[4],
      Wp[9], bp[9], d_out, dec_din[4], dec_dout[4]);
}
